// GPTmuseVAE_88785563943128
// MI455X (gfx1250) — matrix operand order repaired and checked by reading, not compiled, not run on hardware
//
#include <hip/hip_runtime.h>
#include <cstdint>

// ---------------------------------------------------------------------------
// MI455X (gfx1250, wave32) GPT+VAE forward.
// - All heavy GEMMs: v_wmma_f32_16x16x32_bf16 (bf16 A/B, f32 accumulate).
// - Tile staging via ASYNCcnt-tracked global_load_async_to_lds_b128.
// - Attention: one wave per (b, h, 16-query-row tile); scores in LDS,
//   causal softmax in-wave, P@V via WMMA; V chunks async-staged to LDS.
// ---------------------------------------------------------------------------

typedef __attribute__((ext_vector_type(16))) __bf16 v16bf;
typedef __attribute__((ext_vector_type(8)))  float  v8f;

static constexpr int Cdim = 512;   // channels
static constexpr int Tdim = 1024;  // sequence
static constexpr int Bdim = 8;     // batch
static constexpr int Hdim = 8;     // heads
static constexpr int HS   = 64;    // head size
static constexpr int Ldim = 6;     // layers
static constexpr int Rows = Bdim * Tdim;  // 8192

// ---- gfx1250 async global->LDS helpers ------------------------------------
__device__ __forceinline__ unsigned lds_off(const void* p) {
  // generic -> LDS address space: ptrtoint gives raw group-segment offset
  return (unsigned)(size_t)(__attribute__((address_space(3))) const void*)p;
}

template <int OFF>
__device__ __forceinline__ void async_b128(unsigned lds, unsigned long long ga) {
  asm volatile("global_load_async_to_lds_b128 %0, %1, off offset:%c2"
               :: "v"(lds), "v"(ga), "i"(OFF)
               : "memory");
}

__device__ __forceinline__ void wait_async0() {
  asm volatile("s_wait_asynccnt 0x0" ::: "memory");
}

// ======================= embedding =========================================
__global__ void embed_kernel(const int* __restrict__ idx,
                             const float* __restrict__ tok,
                             const float* __restrict__ pos,
                             float* __restrict__ x) {
  size_t gid = (size_t)blockIdx.x * blockDim.x + threadIdx.x;  // < Rows*C
  int c = gid & (Cdim - 1);
  size_t bt = gid >> 9;          // / 512
  int t = bt & (Tdim - 1);
  int tk = idx[bt];
  x[gid] = tok[(size_t)tk * Cdim + c] + pos[(size_t)t * Cdim + c];
}

// ======================= layernorm (wave per row) ==========================
__global__ __launch_bounds__(256)
void ln_kernel(const float* __restrict__ x, const float* __restrict__ g,
               const float* __restrict__ b, __bf16* __restrict__ out) {
  int wv = threadIdx.x >> 5, lane = threadIdx.x & 31;
  size_t row = (size_t)blockIdx.x * 8 + wv;
  const float* xr = x + row * Cdim;
  float vals[16];
  float s = 0.f;
#pragma unroll
  for (int i = 0; i < 16; ++i) { vals[i] = xr[lane + i * 32]; s += vals[i]; }
#pragma unroll
  for (int o = 16; o > 0; o >>= 1) s += __shfl_xor(s, o, 32);
  float mean = s * (1.f / Cdim);
  float vs = 0.f;
#pragma unroll
  for (int i = 0; i < 16; ++i) { float d = vals[i] - mean; vs += d * d; }
#pragma unroll
  for (int o = 16; o > 0; o >>= 1) vs += __shfl_xor(vs, o, 32);
  float rstd = rsqrtf(vs * (1.f / Cdim) + 1e-5f);
#pragma unroll
  for (int i = 0; i < 16; ++i) {
    int c = lane + i * 32;
    out[row * Cdim + c] = (__bf16)((vals[i] - mean) * rstd * g[c] + b[c]);
  }
}

// ======================= WMMA GEMM =========================================
// C[M,N] = A[M,K](bf16) * B[K,N](f32 -> bf16 during LDS staging)
// QKV_B: B indexed as W[h, k, s] with n = h*64+s (no weight repack pass).
// Block = 128 threads (4 waves); 64x64 tile; K-step 32.
// A tile + raw-f32 B tile staged with async global->LDS (ASYNCcnt).
template <bool QKV_B, bool BIAS, bool RELU, bool RESID, bool OUTF, bool OUTB>
__global__ __launch_bounds__(128)
void gemm_wmma(const __bf16* __restrict__ A, const float* __restrict__ B,
               const float* __restrict__ bias, const float* __restrict__ resid,
               float* __restrict__ Cf, __bf16* __restrict__ Cb,
               int M, int N, int K) {
  __shared__ __bf16 sA[64][40];    // [m][k] bf16, padded (80B row, 16B aligned)
  __shared__ __bf16 sBT[64][40];   // [n][k] bf16 (B transposed), padded
  __shared__ float  sBf[32][64];   // [k][n] raw f32 B tile (async landing zone)

  const int tid = threadIdx.x;
  const int wv = tid >> 5, lane = tid & 31;
  const int hf = lane >> 4, l16 = lane & 15;
  const int m0 = blockIdx.y * 64, n0 = blockIdx.x * 64;

  v8f acc[4] = {};

  for (int k0 = 0; k0 < K; k0 += 32) {
    // ---- async stage A tile: 64x32 bf16, 32B per thread ----
    {
      int m = tid >> 1, kk = (tid & 1) * 16;
      const __bf16* src = A + (size_t)(m0 + m) * K + k0 + kk;
      if (k0 + 32 < K) __builtin_prefetch(src + 32, 0, 1);  // global_prefetch
      unsigned la = lds_off(&sA[m][kk]);
      unsigned long long ga = (unsigned long long)(size_t)src;
      async_b128<0>(la, ga);
      async_b128<16>(la, ga);
    }
    // ---- async stage raw B tile: 32x64 f32 (rows 256B contiguous in both
    //      layouts because n0 is 64-aligned), 64B per thread ----
    {
      int kr = tid >> 2;            // B-tile row 0..31
      int q  = (tid & 3) * 16;      // 16-float quarter
      const float* bsrc;
      if constexpr (QKV_B)
        bsrc = B + ((size_t)(n0 >> 6) * K + (k0 + kr)) * 64 + q;
      else
        bsrc = B + (size_t)(k0 + kr) * N + n0 + q;
      unsigned lb = lds_off(&sBf[kr][q]);
      unsigned long long gb = (unsigned long long)(size_t)bsrc;
      async_b128<0>(lb, gb);
      async_b128<16>(lb, gb);
      async_b128<32>(lb, gb);
      async_b128<48>(lb, gb);
    }
    wait_async0();  // own async writes landed (incl. own sBf rows)
    // ---- convert exactly the B elements this thread staged: f32 -> bf16,
    //      transposed into sBT (self-contained, no barrier needed yet) ----
    {
      int kr = tid >> 2, q = (tid & 3) * 16;
#pragma unroll
      for (int i = 0; i < 16; ++i)
        sBT[q + i][kr] = (__bf16)sBf[kr][q + i];
    }
    __syncthreads();  // all waves waited their ASYNCcnt -> tiles visible

    // ---- A fragment: 16x32, lane(l16)=M row; K split by lane half ----
    v16bf af;
    {
      const __bf16* p0 = &sA[wv * 16 + l16][hf * 16];
      const __bf16* p1 = &sA[wv * 16 + l16][hf * 16 + 8];
#pragma unroll
      for (int e = 0; e < 8; ++e) { af[e] = p0[e]; af[8 + e] = p1[e]; }
    }
    // ---- 4 N-tiles of 16; B fragment: 32x16, lane(l16)=N col ----
#pragma unroll
    for (int nt = 0; nt < 4; ++nt) {
      v16bf bfv;
      const __bf16* pb = &sBT[nt * 16 + l16][hf * 16];
#pragma unroll
      for (int e = 0; e < 16; ++e) bfv[e] = pb[e];
      acc[nt] = __builtin_amdgcn_wmma_f32_16x16x32_bf16(
          false, af, false, bfv, (short)0, acc[nt], false, false);
    }
    __syncthreads();
  }

  // ---- epilogue: C/D layout VGPR r -> M = r + 8*half; lane l16 -> N ----
#pragma unroll
  for (int nt = 0; nt < 4; ++nt) {
#pragma unroll
    for (int r = 0; r < 8; ++r) {
      size_t m = m0 + wv * 16 + hf * 8 + r;
      size_t n = n0 + nt * 16 + l16;
      float v = acc[nt][r];
      if constexpr (BIAS) v += bias[n];
      if constexpr (RELU) v = v > 0.f ? v : 0.f;
      if constexpr (RESID) v += resid[m * N + n];
      if constexpr (OUTF) Cf[m * N + n] = v;
      if constexpr (OUTB) Cb[m * N + n] = (__bf16)v;
    }
  }
}

// ======================= attention (wave per 16-row q tile) =================
__global__ __launch_bounds__(32)
void attn_kernel(const __bf16* __restrict__ q, const __bf16* __restrict__ k,
                 const __bf16* __restrict__ v, __bf16* __restrict__ att) {
  __shared__ float  sS[16][1032];  // scores, later holds P (f32), padded
  __shared__ __bf16 sV[32][64];    // async-staged V chunk (rows 128B)

  const int bid = blockIdx.x;            // B*H*(T/16) = 4096
  const int qb = bid & 63;
  const int h  = (bid >> 6) & 7;
  const int b  = bid >> 9;
  const int lane = threadIdx.x;
  const int hf = lane >> 4, l16 = lane & 15;
  const size_t rowbase = (size_t)b * Tdim + qb * 16;

  // Q fragments: two K=32 chunks over HS=64 (loaded once)
  v16bf qf[2];
#pragma unroll
  for (int c = 0; c < 2; ++c) {
    const __bf16* p = q + (rowbase + l16) * Cdim + h * HS + c * 32;
#pragma unroll
    for (int e = 0; e < 8; ++e) {
      qf[c][e]     = p[hf * 16 + e];
      qf[c][8 + e] = p[hf * 16 + 8 + e];
    }
  }

  // ---- scores: S = (Q K^T) * scale, causal mask, into LDS ----
  for (int jt = 0; jt <= qb; ++jt) {
    v8f acc = {};
#pragma unroll
    for (int c = 0; c < 2; ++c) {
      v16bf kf;
      const __bf16* p =
          k + ((size_t)b * Tdim + jt * 16 + l16) * Cdim + h * HS + c * 32 + hf * 16;
#pragma unroll
      for (int e = 0; e < 16; ++e) kf[e] = p[e];
      acc = __builtin_amdgcn_wmma_f32_16x16x32_bf16(
          false, qf[c], false, kf, (short)0, acc, false, false);
    }
    int u = jt * 16 + l16;
#pragma unroll
    for (int r = 0; r < 8; ++r) {
      int m = qb * 16 + hf * 8 + r;
      sS[hf * 8 + r][u] = (u <= m) ? acc[r] * 0.125f : -__builtin_inff();
    }
  }

  const int limit = (qb + 1) * 16;
  const int chunks = (limit + 31) >> 5;
  const int climit = chunks * 32;

  // ---- softmax: 2 lanes per row (row = l16, phase = hf), in-place P ----
  {
    int r = l16;
    float mx = -__builtin_inff();
    for (int u = hf; u < limit; u += 2) mx = fmaxf(mx, sS[r][u]);
    mx = fmaxf(mx, __shfl_xor(mx, 16, 32));
    float sum = 0.f;
    for (int u = hf; u < limit; u += 2) sum += __expf(sS[r][u] - mx);
    sum += __shfl_xor(sum, 16, 32);
    float inv = 1.f / sum;
    for (int u = hf; u < limit; u += 2) sS[r][u] = __expf(sS[r][u] - mx) * inv;
    for (int u = limit + hf; u < climit; u += 2) sS[r][u] = 0.f;  // pad chunk
  }

  // ---- O = P @ V via WMMA; V chunk async-staged to LDS each iteration ----
  v8f oacc[4] = {};
  for (int c = 0; c < chunks; ++c) {
    {
      const __bf16* gsrc =
          v + ((size_t)b * Tdim + c * 32 + lane) * Cdim + h * HS;
      unsigned la = lds_off(&sV[lane][0]);
      unsigned long long ga = (unsigned long long)(size_t)gsrc;
      async_b128<0>(la, ga);   async_b128<16>(la, ga);
      async_b128<32>(la, ga);  async_b128<48>(la, ga);
      async_b128<64>(la, ga);  async_b128<80>(la, ga);
      async_b128<96>(la, ga);  async_b128<112>(la, ga);
      wait_async0();
    }
    v16bf pf;
    const float* pp = &sS[l16][c * 32];
#pragma unroll
    for (int e = 0; e < 8; ++e) {
      pf[e]     = (__bf16)pp[hf * 16 + e];
      pf[8 + e] = (__bf16)pp[hf * 16 + 8 + e];
    }
#pragma unroll
    for (int nt = 0; nt < 4; ++nt) {
      v16bf vf;
#pragma unroll
      for (int e = 0; e < 16; ++e)
        vf[e] = sV[hf * 16 + e][nt * 16 + l16];
      oacc[nt] = __builtin_amdgcn_wmma_f32_16x16x32_bf16(
          false, pf, false, vf, (short)0, oacc[nt], false, false);
    }
  }

#pragma unroll
  for (int nt = 0; nt < 4; ++nt)
#pragma unroll
    for (int r = 0; r < 8; ++r) {
      size_t row = rowbase + hf * 8 + r;
      att[row * Cdim + h * HS + nt * 16 + l16] = (__bf16)oacc[nt][r];
    }
}

// ======================= VAE bottleneck (tiny mats, per-row) ================
__global__ __launch_bounds__(64)
void vae_kernel(const __bf16* __restrict__ xf, const float* __restrict__ eps,
                const float* __restrict__ w1, const float* __restrict__ b1,
                const float* __restrict__ w21, const float* __restrict__ b21,
                const float* __restrict__ w22, const float* __restrict__ b22,
                const float* __restrict__ w3, const float* __restrict__ b3,
                const float* __restrict__ w4, const float* __restrict__ b4,
                __bf16* __restrict__ dec) {
  const int row = blockIdx.x;
  const int t = threadIdx.x;
  __shared__ float xr[512];
  __shared__ float h1[32];
  __shared__ float mu_s[16], lv_s[16];
  __shared__ float z[16];
  __shared__ float h3[64];

  for (int i = t; i < 512; i += 64) xr[i] = (float)xf[(size_t)row * 512 + i];
  __syncthreads();
  if (t < 32) {
    float s = b1[t];
    for (int c = 0; c < 512; ++c) s += xr[c] * w1[c * 32 + t];
    h1[t] = s > 0.f ? s : 0.f;
  }
  __syncthreads();
  if (t < 32) {
    int j = t & 15;
    const float* w = (t < 16) ? w21 : w22;
    const float* bb = (t < 16) ? b21 : b22;
    float s = bb[j];
    for (int c = 0; c < 32; ++c) s += h1[c] * w[c * 16 + j];
    if (t < 16) mu_s[j] = s; else lv_s[j] = s;
  }
  __syncthreads();
  if (t < 16) z[t] = mu_s[t] + eps[(size_t)row * 16 + t] * __expf(0.5f * lv_s[t]);
  __syncthreads();
  {
    float s = b3[t];
    for (int c = 0; c < 16; ++c) s += z[c] * w3[c * 64 + t];
    h3[t] = s > 0.f ? s : 0.f;
  }
  __syncthreads();
  for (int o = t; o < 512; o += 64) {
    float s = b4[o];
    for (int c = 0; c < 64; ++c) s += h3[c] * w4[c * 512 + o];
    dec[(size_t)row * 512 + o] = (__bf16)(1.f / (1.f + __expf(-s)));
  }
}

// ======================= host orchestration =================================
extern "C" void kernel_launch(void* const* d_in, const int* in_sizes, int n_in,
                              void* d_out, int out_size, void* d_ws,
                              size_t ws_size, hipStream_t stream) {
  const int*   idx   = (const int*)d_in[0];
  const float* eps   = (const float*)d_in[1];
  const float* tok   = (const float*)d_in[2];
  const float* pos   = (const float*)d_in[3];
  const float* Wq    = (const float*)d_in[4];
  const float* Wk    = (const float*)d_in[5];
  const float* Wv    = (const float*)d_in[6];
  const float* Wproj = (const float*)d_in[7];
  const float* bproj = (const float*)d_in[8];
  const float* ln1g  = (const float*)d_in[9];
  const float* ln1b  = (const float*)d_in[10];
  const float* ln2g  = (const float*)d_in[11];
  const float* ln2b  = (const float*)d_in[12];
  const float* Wff1  = (const float*)d_in[13];
  const float* bff1  = (const float*)d_in[14];
  const float* Wff2  = (const float*)d_in[15];
  const float* bff2  = (const float*)d_in[16];
  const float* lnfg  = (const float*)d_in[17];
  const float* lnfb  = (const float*)d_in[18];
  const float* w1    = (const float*)d_in[19];
  const float* b1    = (const float*)d_in[20];
  const float* w21   = (const float*)d_in[21];
  const float* b21   = (const float*)d_in[22];
  const float* w22   = (const float*)d_in[23];
  const float* b22   = (const float*)d_in[24];
  const float* w3    = (const float*)d_in[25];
  const float* b3    = (const float*)d_in[26];
  const float* w4    = (const float*)d_in[27];
  const float* b4    = (const float*)d_in[28];
  const float* lmw   = (const float*)d_in[29];
  const float* lmb   = (const float*)d_in[30];

  // workspace carve-up
  char* p = (char*)d_ws;
  auto alloc = [&](size_t bytes) {
    char* r = p; p += (bytes + 255) & ~size_t(255); return r;
  };
  float*  x    = (float*)alloc((size_t)Rows * Cdim * 4);
  __bf16* hbuf = (__bf16*)alloc((size_t)Rows * Cdim * 2);
  __bf16* qbuf = (__bf16*)alloc((size_t)Rows * Cdim * 2);
  __bf16* kbuf = (__bf16*)alloc((size_t)Rows * Cdim * 2);
  __bf16* vbuf = (__bf16*)alloc((size_t)Rows * Cdim * 2);
  __bf16* abuf = (__bf16*)alloc((size_t)Rows * Cdim * 2);
  __bf16* fbuf = (__bf16*)alloc((size_t)Rows * 4 * Cdim * 2);
  __bf16* dbuf = (__bf16*)alloc((size_t)Rows * Cdim * 2);

  const dim3 gC(Cdim / 64, Rows / 64);       // N=512 GEMMs
  const dim3 gF(4 * Cdim / 64, Rows / 64);   // N=2048 GEMM

  embed_kernel<<<(Rows * Cdim) / 256, 256, 0, stream>>>(idx, tok, pos, x);

  for (int l = 0; l < Ldim; ++l) {
    ln_kernel<<<Rows / 8, 256, 0, stream>>>(x, ln1g + l * Cdim, ln1b + l * Cdim, hbuf);

    const size_t wqoff = (size_t)l * Hdim * Cdim * HS;
    gemm_wmma<true, false, false, false, false, true><<<gC, 128, 0, stream>>>(
        hbuf, Wq + wqoff, nullptr, nullptr, nullptr, qbuf, Rows, Cdim, Cdim);
    gemm_wmma<true, false, false, false, false, true><<<gC, 128, 0, stream>>>(
        hbuf, Wk + wqoff, nullptr, nullptr, nullptr, kbuf, Rows, Cdim, Cdim);
    gemm_wmma<true, false, false, false, false, true><<<gC, 128, 0, stream>>>(
        hbuf, Wv + wqoff, nullptr, nullptr, nullptr, vbuf, Rows, Cdim, Cdim);

    attn_kernel<<<Bdim * Hdim * (Tdim / 16), 32, 0, stream>>>(qbuf, kbuf, vbuf, abuf);

    gemm_wmma<false, true, false, true, true, false><<<gC, 128, 0, stream>>>(
        abuf, Wproj + (size_t)l * Cdim * Cdim, bproj + l * Cdim, x, x, nullptr,
        Rows, Cdim, Cdim);

    ln_kernel<<<Rows / 8, 256, 0, stream>>>(x, ln2g + l * Cdim, ln2b + l * Cdim, hbuf);

    gemm_wmma<false, true, true, false, false, true><<<gF, 128, 0, stream>>>(
        hbuf, Wff1 + (size_t)l * Cdim * 4 * Cdim, bff1 + l * 4 * Cdim, nullptr,
        nullptr, fbuf, Rows, 4 * Cdim, Cdim);
    gemm_wmma<false, true, false, true, true, false><<<gC, 128, 0, stream>>>(
        fbuf, Wff2 + (size_t)l * 4 * Cdim * Cdim, bff2 + l * Cdim, x, x, nullptr,
        Rows, Cdim, 4 * Cdim);
  }

  ln_kernel<<<Rows / 8, 256, 0, stream>>>(x, lnfg, lnfb, hbuf);

  vae_kernel<<<Rows, 64, 0, stream>>>(hbuf, eps, w1, b1, w21, b21, w22, b22,
                                      w3, b3, w4, b4, dbuf);

  gemm_wmma<false, true, false, false, true, false><<<gC, 128, 0, stream>>>(
      dbuf, lmw, lmb, nullptr, (float*)d_out, nullptr, Rows, Cdim, Cdim);
}
